// XLSTMClassifier_49469433315482
// MI455X (gfx1250) — matrix operand order repaired and checked by reading, not compiled, not run on hardware
//
#include <hip/hip_runtime.h>
#include <cstdint>
#include <cstddef>

#define S_LEN 1500
#define D_MODEL 256
#define DI 512
#define NHM 4
#define DHM 128
#define BATCH 8
#define FF_DIM 332
#define NROWS (BATCH * S_LEN)   // 12000
#define SPAD 1504               // 94*16
#define NTQ 94                  // ceil(S/16)

typedef __bf16 v16bf __attribute__((ext_vector_type(16)));
typedef float  v8f   __attribute__((ext_vector_type(8)));

union AFrag { uint4 u[2]; v16bf v; };

__device__ inline v8f wmma_bf16(v16bf a, v16bf b, v8f c) {
  return __builtin_amdgcn_wmma_f32_16x16x32_bf16(false, a, false, b, (short)0, c, false, false);
}
__device__ inline unsigned short f2bf(float f) {
  unsigned int u = __float_as_uint(f);
  u = (u + 0x7FFFu + ((u >> 16) & 1u)) >> 16;
  return (unsigned short)u;
}
__device__ inline unsigned int pack2bf(float lo, float hi) {
  return (unsigned int)f2bf(lo) | ((unsigned int)f2bf(hi) << 16);
}
__device__ inline float logsig(float x) { return fminf(x, 0.f) - log1pf(__expf(-fabsf(x))); }
__device__ inline float silu_f(float x) { return x / (1.f + __expf(-x)); }
__device__ inline float gelu_f(float x) {
  float x3 = x * x * x;
  return 0.5f * x * (1.f + tanhf(0.7978845608028654f * (x + 0.044715f * x3)));
}
__device__ inline float wred32(float v) {
#pragma unroll
  for (int m = 16; m > 0; m >>= 1) v += __shfl_xor(v, m, 32);
  return v;
}
__device__ inline int imin(int a, int b) { return a < b ? a : b; }
__device__ inline int imax(int a, int b) { return a > b ? a : b; }

// ---------------------------------------------------------------------------
// Generic bf16 WMMA GEMM: C[M,N] = A[M,K] * B[K,N] + bias (+ residual)
// Block = 128 thr (4 waves), tile 64x64, K-step 32. f32 in, bf16 staged in LDS
// as packed u32 pairs. Uniform-branch fast path (vector loads) for interior
// tiles; branchless clamped edge path otherwise.
// ---------------------------------------------------------------------------
__global__ __launch_bounds__(128) void gemm_bf16_kernel(
    const float* __restrict__ A, int lda,
    const float* __restrict__ Bm, int ldb,
    const float* __restrict__ bias,
    const float* __restrict__ res,
    float* __restrict__ C, int ldc,
    int M, int N, int K)
{
  __shared__ alignas(16) unsigned int As32[64][16];  // [m][k/2], 2 bf16 / word
  __shared__ alignas(16) unsigned int Bs32[64][16];  // [n][k/2] (B transposed)
  const int tid  = threadIdx.x;
  const int lane = tid & 31;
  const int wave = tid >> 5;
  const int bm0  = blockIdx.y * 64;
  const int bn0  = blockIdx.x * 64;
  const int mw   = (wave & 1) * 32;
  const int nw   = (wave >> 1) * 32;
  const int lr   = lane & 15;
  const int khf  = lane >> 4;
  const bool interior = (bm0 + 64 <= M) && (bn0 + 64 <= N);

  v8f acc[2][2];
  acc[0][0] = {}; acc[0][1] = {}; acc[1][0] = {}; acc[1][1] = {};

  for (int k0 = 0; k0 < K; k0 += 32) {
    if (interior && (k0 + 32 <= K)) {
      // ---- fast path: float2 loads, packed bf16 stores ----
#pragma unroll
      for (int i = tid; i < 1024; i += 128) {        // A: 64 x 16 pairs
        int m = i >> 4, k2 = i & 15;
        float2 f = *(const float2*)(A + (size_t)(bm0 + m) * lda + k0 + 2 * k2);
        As32[m][k2] = pack2bf(f.x, f.y);
      }
#pragma unroll
      for (int i = tid; i < 1024; i += 128) {        // B: 16 kpairs x 64 n
        int k2 = i >> 6, n = i & 63;
        float lo = Bm[(size_t)(k0 + 2 * k2) * ldb + bn0 + n];
        float hi = Bm[(size_t)(k0 + 2 * k2 + 1) * ldb + bn0 + n];
        Bs32[n][k2] = pack2bf(lo, hi);
      }
      if (k0 + 64 <= K) {   // prefetch next K tile (global_prefetch_b8)
        __builtin_prefetch(A + (size_t)(bm0 + (tid & 63)) * lda + k0 + 32, 0, 0);
        __builtin_prefetch(Bm + (size_t)(k0 + 32 + (tid >> 2)) * ldb + bn0, 0, 0);
      }
    } else {
      // ---- edge path: branchless clamped loads + select ----
      unsigned short* As16 = (unsigned short*)As32;
      unsigned short* Bs16 = (unsigned short*)Bs32;
#pragma unroll
      for (int i = tid; i < 2048; i += 128) {
        int m = i >> 5, k = i & 31;
        int gm = imin(bm0 + m, M - 1), gkk = imin(k0 + k, K - 1);
        float v = A[(size_t)gm * lda + gkk];
        v = (bm0 + m < M && k0 + k < K) ? v : 0.f;
        As16[m * 32 + k] = f2bf(v);
      }
#pragma unroll
      for (int i = tid; i < 2048; i += 128) {
        int k = i >> 6, n = i & 63;
        int gkk = imin(k0 + k, K - 1), gn = imin(bn0 + n, N - 1);
        float v = Bm[(size_t)gkk * ldb + gn];
        v = (k0 + k < K && bn0 + n < N) ? v : 0.f;
        Bs16[n * 32 + k] = f2bf(v);
      }
    }
    __syncthreads();

    AFrag a[2];
#pragma unroll
    for (int mi = 0; mi < 2; ++mi) {
      const uint4* r4 = (const uint4*)&As32[mw + mi * 16 + lr][0];
      a[mi].u[0] = r4[2 * khf];
      a[mi].u[1] = r4[2 * khf + 1];
    }
    v16bf bf[2];
#pragma unroll
    for (int ni = 0; ni < 2; ++ni) {
      const unsigned int* rp = &Bs32[nw + ni * 16 + lr][0];
      bf[ni] = *(const v16bf*)(rp + khf * 8);
    }
#pragma unroll
    for (int mi = 0; mi < 2; ++mi)
#pragma unroll
      for (int ni = 0; ni < 2; ++ni)
        acc[mi][ni] = wmma_bf16(a[mi].v, bf[ni], acc[mi][ni]);
    __syncthreads();
  }

#pragma unroll
  for (int mi = 0; mi < 2; ++mi)
#pragma unroll
    for (int ni = 0; ni < 2; ++ni)
#pragma unroll
      for (int r = 0; r < 8; ++r) {
        int row = bm0 + mw + mi * 16 + r + 8 * khf;
        int col = bn0 + nw + ni * 16 + lr;
        if (row < M && col < N) {
          float v = acc[mi][ni][r] + (bias ? bias[col] : 0.f);
          if (res) v += res[(size_t)row * ldc + col];
          C[(size_t)row * ldc + col] = v;
        }
      }
}

// ---------------------------------------------------------------------------
// Embedding gather + layernorm (rows of 256); wave-butterfly reductions.
// ---------------------------------------------------------------------------
__global__ __launch_bounds__(256) void embed_ln_kernel(
    const int* __restrict__ seq, const float* __restrict__ embed,
    const float* __restrict__ w, float* __restrict__ x, float* __restrict__ xln)
{
  __shared__ float r1[8], r2[8];
  int row = blockIdx.x, t = threadIdx.x;
  int wv = t >> 5, lane = t & 31;
  int tok = seq[row];
  float v = embed[(size_t)tok * D_MODEL + t];
  x[(size_t)row * D_MODEL + t] = v;
  float a = wred32(v), bq = wred32(v * v);
  if (lane == 0) { r1[wv] = a; r2[wv] = bq; }
  __syncthreads();
  float s1 = 0.f, s2 = 0.f;
#pragma unroll
  for (int i = 0; i < 8; ++i) { s1 += r1[i]; s2 += r2[i]; }
  float mean = s1 / 256.f;
  float var = s2 / 256.f - mean * mean;
  xln[(size_t)row * D_MODEL + t] = (v - mean) * rsqrtf(var + 1e-5f) * w[t];
}

__global__ __launch_bounds__(256) void ln256_kernel(
    const float* __restrict__ in, const float* __restrict__ w, float* __restrict__ out)
{
  __shared__ float r1[8], r2[8];
  int row = blockIdx.x, t = threadIdx.x;
  int wv = t >> 5, lane = t & 31;
  float v = in[(size_t)row * D_MODEL + t];
  float a = wred32(v), bq = wred32(v * v);
  if (lane == 0) { r1[wv] = a; r2[wv] = bq; }
  __syncthreads();
  float s1 = 0.f, s2 = 0.f;
#pragma unroll
  for (int i = 0; i < 8; ++i) { s1 += r1[i]; s2 += r2[i]; }
  float mean = s1 / 256.f;
  float var = s2 / 256.f - mean * mean;
  out[(size_t)row * D_MODEL + t] = (v - mean) * rsqrtf(var + 1e-5f) * w[t];
}

// ---------------------------------------------------------------------------
// Depthwise causal conv (K=4) + SiLU (branchless taps).
// ---------------------------------------------------------------------------
__global__ __launch_bounds__(256) void mconv_kernel(
    const float* __restrict__ up, const float* __restrict__ cw,
    const float* __restrict__ cb, float* __restrict__ xc)
{
  int idx = blockIdx.x * 256 + threadIdx.x;
  if (idx >= NROWS * DI) return;
  int row = idx >> 9, c = idx & 511;
  int b = row / S_LEN, s = row % S_LEN;
  float acc = cb[c];
#pragma unroll
  for (int t = 0; t < 4; ++t) {
    int sp = s - 3 + t;
    float xv = up[((size_t)(b * S_LEN + imax(sp, 0))) * (2 * DI) + c];
    acc += (sp >= 0 ? xv : 0.f) * cw[t * DI + c];
  }
  xc[idx] = silu_f(acc);
}

__global__ __launch_bounds__(256) void sconv_kernel(
    const float* __restrict__ xin, const float* __restrict__ cw,
    const float* __restrict__ cb, float* __restrict__ xc)
{
  int idx = blockIdx.x * 256 + threadIdx.x;
  if (idx >= NROWS * D_MODEL) return;
  int row = idx >> 8, c = idx & 255;
  int b = row / S_LEN, s = row % S_LEN;
  float acc = cb[c];
#pragma unroll
  for (int t = 0; t < 4; ++t) {
    int sp = s - 3 + t;
    float xv = xin[((size_t)(b * S_LEN + imax(sp, 0))) * D_MODEL + c];
    acc += (sp >= 0 ? xv : 0.f) * cw[t * D_MODEL + c];
  }
  xc[idx] = silu_f(acc);
}

// ---------------------------------------------------------------------------
// Block-diagonal qkv (4x4 blocks): q,k from xc; v from xi. Writes f32 + bf16.
// ---------------------------------------------------------------------------
__global__ __launch_bounds__(256) void qkv_kernel(
    const float* __restrict__ xc, const float* __restrict__ up,
    const float* __restrict__ Wq, const float* __restrict__ Wk, const float* __restrict__ Wv,
    float* __restrict__ q, float* __restrict__ k, float* __restrict__ v,
    unsigned short* __restrict__ qb, unsigned short* __restrict__ kb)
{
  int idx = blockIdx.x * 256 + threadIdx.x;
  if (idx >= NROWS * 128) return;
  int row = idx >> 7, nb = idx & 127;
  float xc4[4], xi4[4];
#pragma unroll
  for (int i = 0; i < 4; ++i) {
    xc4[i] = xc[(size_t)row * DI + nb * 4 + i];
    xi4[i] = up[(size_t)row * (2 * DI) + nb * 4 + i];
  }
#pragma unroll
  for (int o = 0; o < 4; ++o) {
    float qs = 0.f, ks = 0.f, vs = 0.f;
#pragma unroll
    for (int i = 0; i < 4; ++i) {
      qs += xc4[i] * Wq[nb * 16 + i * 4 + o];
      ks += xc4[i] * Wk[nb * 16 + i * 4 + o];
      vs += xi4[i] * Wv[nb * 16 + i * 4 + o];
    }
    size_t oi = (size_t)row * DI + nb * 4 + o;
    q[oi] = qs; k[oi] = ks; v[oi] = vs;
    qb[oi] = f2bf(qs);
    kb[oi] = f2bf(ks * 0.08838834764831845f);  // 1/sqrt(128)
  }
}

// v -> vT (b,h,d,s) bf16, s padded to SPAD with zeros
__global__ __launch_bounds__(256) void vtrans_kernel(
    const float* __restrict__ v, unsigned short* __restrict__ vT)
{
  int idx = blockIdx.x * 256 + threadIdx.x;
  if (idx >= BATCH * NHM * DHM * SPAD) return;
  int sp = idx % SPAD;
  int rest = idx / SPAD;
  int d = rest % DHM;
  int bh = rest / DHM;
  int b = bh >> 2, h = bh & 3;
  float val = 0.f;
  if (sp < S_LEN) val = v[((size_t)(b * S_LEN + sp)) * DI + h * DHM + d];
  vT[idx] = f2bf(val);
}

// ---------------------------------------------------------------------------
// ipre/fpre = [q|k|v] @ Wig/Wfg + bias, stored (b,h,s)
// ---------------------------------------------------------------------------
__global__ __launch_bounds__(256) void gatepre_kernel(
    const float* __restrict__ q, const float* __restrict__ k, const float* __restrict__ v,
    const float* __restrict__ Wig, const float* __restrict__ big,
    const float* __restrict__ Wfg, const float* __restrict__ bfg,
    float* __restrict__ ipre, float* __restrict__ fpre)
{
  int row = blockIdx.x;
  int b = row / S_LEN, s = row % S_LEN;
  int wave = threadIdx.x >> 5, lane = threadIdx.x & 31;
  int gate = wave >> 2, head = wave & 3;
  const float* W = gate ? Wfg : Wig;
  float sum = 0.f;
  for (int j = lane; j < 3 * DI; j += 32) {
    float xv = (j < DI) ? q[(size_t)row * DI + j]
             : (j < 2 * DI) ? k[(size_t)row * DI + j - DI]
             : v[(size_t)row * DI + j - 2 * DI];
    sum += xv * W[(size_t)j * NHM + head];
  }
  sum = wred32(sum);
  if (lane == 0) {
    size_t oi = ((size_t)(b * NHM + head)) * S_LEN + s;
    if (gate == 0) ipre[oi] = sum + big[head];
    else           fpre[oi] = sum + bfg[head];
  }
}

// serial decay scan per (b,h): cum = cumsum(logsig(fpre)); gk = ipre - cum;
// pm = prefix-max(gk)
__global__ __launch_bounds__(32) void decay_scan_kernel(
    const float* __restrict__ ipre, const float* __restrict__ fpre,
    float* __restrict__ cum, float* __restrict__ gk, float* __restrict__ pm)
{
  int t = threadIdx.x;  // (b,h)
  if (t >= BATCH * NHM) return;
  float c = 0.f, pmax = -3.0e38f;
  for (int s = 0; s < S_LEN; ++s) {
    size_t i = (size_t)t * S_LEN + s;
    c += logsig(fpre[i]);
    cum[i] = c;
    float g = ipre[i] - c;
    gk[i] = g;
    pmax = fmaxf(pmax, g);
    pm[i] = pmax;
  }
}

// ---------------------------------------------------------------------------
// Flash-style mLSTM: one wave per (b,h,qtile of 16 rows).
// ---------------------------------------------------------------------------
__global__ __launch_bounds__(32) void mlstm_attn_kernel(
    const unsigned short* __restrict__ qb, const unsigned short* __restrict__ kb,
    const unsigned short* __restrict__ vT,
    const float* __restrict__ gk, const float* __restrict__ pm,
    const float* __restrict__ cum, float* __restrict__ hout)
{
  __shared__ alignas(16) unsigned short sA[16][32];
  const int lane = threadIdx.x;
  const int lr = lane & 15, khf = lane >> 4;
  int bid = blockIdx.x;
  const int qt = bid % NTQ; bid /= NTQ;
  const int h = bid % NHM;
  const int b = bid / NHM;
  const int qbase = qt * 16;
  const int bh = b * NHM + h;

  // preload q tile A-fragments (4 K-chunks of 32 over d=128)
  AFrag qa[4];
  {
    int qrow = imin(qbase + lr, S_LEN - 1);
    const uint4* qp = (const uint4*)(qb + ((size_t)(b * S_LEN + qrow)) * DI + h * DHM);
#pragma unroll
    for (int c = 0; c < 4; ++c) {
      qa[c].u[0] = qp[4 * c + 2 * khf];
      qa[c].u[1] = qp[4 * c + 2 * khf + 1];
    }
  }
  float pmv[8], maxd[8];
#pragma unroll
  for (int r = 0; r < 8; ++r) {
    int qrow = imin(qbase + r + 8 * khf, S_LEN - 1);
    pmv[r]  = pm[(size_t)bh * S_LEN + qrow];
    maxd[r] = cum[(size_t)bh * S_LEN + qrow] + pmv[r];
  }

  v8f hacc[8];
#pragma unroll
  for (int dc = 0; dc < 8; ++dc) hacc[dc] = {};
  float rsum[8] = {0.f, 0.f, 0.f, 0.f, 0.f, 0.f, 0.f, 0.f};

  const int kend = imin(qbase + 16, S_LEN);
  for (int kb0 = 0; kb0 < kend; kb0 += 32) {
    // prefetch next k-block rows of k and vT
    int knext = kb0 + 32;
    if (knext < kend) {
      __builtin_prefetch(kb + ((size_t)(b * S_LEN + imin(knext + lane, S_LEN - 1))) * DI + h * DHM, 0, 0);
      __builtin_prefetch(vT + ((size_t)bh * DHM + 4 * lane) * SPAD + knext, 0, 0);
    }
#pragma unroll
    for (int sub = 0; sub < 2; ++sub) {
      const int kt = kb0 + sub * 16;
      int krow = imin(kt + lr, S_LEN - 1);
      const unsigned short* kp = kb + ((size_t)(b * S_LEN + krow)) * DI + h * DHM;
      v8f sacc = {};
#pragma unroll
      for (int c = 0; c < 4; ++c) {
        v16bf bf = *(const v16bf*)(kp + c * 32 + khf * 16);
        sacc = wmma_bf16(qa[c].v, bf, sacc);
      }
      const int kglob = kt + lr;                      // C-frag column index
      float gkl = gk[(size_t)bh * S_LEN + imin(kglob, S_LEN - 1)];
#pragma unroll
      for (int r = 0; r < 8; ++r) {
        int qglob = qbase + r + 8 * khf;
        float w = (kglob <= qglob && kglob < S_LEN && qglob < S_LEN)
                    ? __expf(gkl - pmv[r]) : 0.f;
        float sv = sacc[r] * w;
        rsum[r] += sv;
        sA[r + 8 * khf][lr + 16 * sub] = f2bf(sv);
      }
    }
    __syncthreads();
    AFrag sa;
    {
      const uint4* sp = (const uint4*)&sA[lr][0];
      sa.u[0] = sp[2 * khf];
      sa.u[1] = sp[2 * khf + 1];
    }
    __syncthreads();
#pragma unroll
    for (int dc = 0; dc < 8; ++dc) {
      const unsigned short* vp =
          vT + ((size_t)bh * DHM + dc * 16 + lr) * SPAD + kb0 + khf * 16;
      v16bf bf = *(const v16bf*)vp;
      hacc[dc] = wmma_bf16(sa.v, bf, hacc[dc]);
    }
  }

  // reduce row sums across the 16 lanes of each half
#pragma unroll
  for (int r = 0; r < 8; ++r) {
    float s = rsum[r];
#pragma unroll
    for (int m = 1; m < 16; m <<= 1) s += __shfl_xor(s, m, 32);
    rsum[r] = s;
  }
#pragma unroll
  for (int r = 0; r < 8; ++r) {
    int qglob = qbase + r + 8 * khf;
    if (qglob < S_LEN) {
      float norm = fmaxf(fabsf(rsum[r]), __expf(-maxd[r])) + 1e-6f;
      float inv = 1.f / norm;
#pragma unroll
      for (int dc = 0; dc < 8; ++dc)
        hout[((size_t)(b * S_LEN + qglob)) * DI + h * DHM + dc * 16 + lr] =
            hacc[dc][r] * inv;
    }
  }
}

// ---------------------------------------------------------------------------
// mLSTM epilogue: head-norm + skip + silu(z) gate -> hgate (f32)
// ---------------------------------------------------------------------------
__global__ __launch_bounds__(128) void mlstm_out_kernel(
    const float* __restrict__ hm, const float* __restrict__ mhn_w,
    const float* __restrict__ skip, const float* __restrict__ xc,
    const float* __restrict__ up, float* __restrict__ hgate)
{
  int row = blockIdx.x;
  int h = threadIdx.x >> 5, lane = threadIdx.x & 31;
  float vals[4], s1 = 0.f, s2 = 0.f;
#pragma unroll
  for (int j = 0; j < 4; ++j) {
    vals[j] = hm[(size_t)row * DI + h * DHM + lane * 4 + j];
    s1 += vals[j]; s2 += vals[j] * vals[j];
  }
  s1 = wred32(s1); s2 = wred32(s2);
  float mean = s1 / (float)DHM;
  float var = s2 / (float)DHM - mean * mean;
  float inv = rsqrtf(var + 1e-5f);
#pragma unroll
  for (int j = 0; j < 4; ++j) {
    int c = h * DHM + lane * 4 + j;
    float hn = (vals[j] - mean) * inv * mhn_w[c];
    float hs = hn + skip[c] * xc[(size_t)row * DI + c];
    float zv = up[(size_t)row * (2 * DI) + DI + c];
    hgate[(size_t)row * DI + c] = hs * silu_f(zv);
  }
}

// ---------------------------------------------------------------------------
// sLSTM recurrent scan: one block per (b,head), 64 threads (element each),
// R matrices cached in dynamic LDS; next-step gates prefetched.
// ---------------------------------------------------------------------------
__global__ __launch_bounds__(64) void slstm_scan_kernel(
    const float* __restrict__ gi, const float* __restrict__ gf,
    const float* __restrict__ gz, const float* __restrict__ go,
    const float* __restrict__ Ri, const float* __restrict__ Rf,
    const float* __restrict__ Rz, const float* __restrict__ Ro,
    float* __restrict__ hs_out)
{
  extern __shared__ float sm[];
  float* sRi = sm;
  float* sRf = sm + 4096;
  float* sRz = sm + 8192;
  float* sRo = sm + 12288;
  float* sh  = sm + 16384;
  int bh = blockIdx.x;
  int b = bh >> 2, h = bh & 3;
  int e = threadIdx.x;
  for (int i = e; i < 4096; i += 64) {
    sRi[i] = Ri[(size_t)h * 4096 + i];
    sRf[i] = Rf[(size_t)h * 4096 + i];
    sRz[i] = Rz[(size_t)h * 4096 + i];
    sRo[i] = Ro[(size_t)h * 4096 + i];
  }
  sh[e] = 0.f;
  float c = 0.f, n = 0.f, m = 0.f;
  __syncthreads();
  for (int s = 0; s < S_LEN; ++s) {
    size_t gidx = ((size_t)(b * S_LEN + s)) * D_MODEL + h * 64 + e;
    if (s + 1 < S_LEN) {   // prefetch next timestep's gate rows
      __builtin_prefetch(gi + gidx + D_MODEL, 0, 0);
      __builtin_prefetch(gf + gidx + D_MODEL, 0, 0);
      __builtin_prefetch(gz + gidx + D_MODEL, 0, 0);
      __builtin_prefetch(go + gidx + D_MODEL, 0, 0);
    }
    float ir = gi[gidx], fr = gf[gidx], zr = gz[gidx], orr = go[gidx];
    for (int d = 0; d < 64; ++d) {
      float hd = sh[d];
      ir  += hd * sRi[d * 64 + e];
      fr  += hd * sRf[d * 64 + e];
      zr  += hd * sRz[d * 64 + e];
      orr += hd * sRo[d * 64 + e];
    }
    float lfm = m + logsig(fr);
    float mnew = fmaxf(ir, lfm);
    float iexp = __expf(ir - mnew);
    float fexp = __expf(lfm - mnew);
    c = fexp * c + iexp * tanhf(zr);
    n = fexp * n + iexp;
    float hnew = (1.f / (1.f + __expf(-orr))) * c / (n + 1e-6f);
    m = mnew;
    hs_out[gidx] = hnew;
    __syncthreads();
    sh[e] = hnew;
    __syncthreads();
  }
}

// sLSTM head-norm (Dh=64) + residual: x3 = x2 + gn(hs)*w
__global__ __launch_bounds__(128) void slstm_out_kernel(
    const float* __restrict__ hs, const float* __restrict__ gn_w,
    const float* __restrict__ x2, float* __restrict__ x3)
{
  int row = blockIdx.x;
  int h = threadIdx.x >> 5, lane = threadIdx.x & 31;
  float vals[2], s1 = 0.f, s2 = 0.f;
#pragma unroll
  for (int j = 0; j < 2; ++j) {
    vals[j] = hs[(size_t)row * D_MODEL + h * 64 + lane * 2 + j];
    s1 += vals[j]; s2 += vals[j] * vals[j];
  }
  s1 = wred32(s1); s2 = wred32(s2);
  float mean = s1 / 64.f;
  float var = s2 / 64.f - mean * mean;
  float inv = rsqrtf(var + 1e-5f);
#pragma unroll
  for (int j = 0; j < 2; ++j) {
    int cidx = h * 64 + lane * 2 + j;
    size_t oi = (size_t)row * D_MODEL + cidx;
    x3[oi] = x2[oi] + (vals[j] - mean) * inv * gn_w[cidx];
  }
}

__global__ __launch_bounds__(256) void ffn_act_kernel(
    const float* __restrict__ g, const float* __restrict__ u,
    float* __restrict__ hx, int n)
{
  int idx = blockIdx.x * 256 + threadIdx.x;
  if (idx < n) hx[idx] = gelu_f(g[idx]) * u[idx];
}

__global__ __launch_bounds__(256) void pool_fc_kernel(
    const float* __restrict__ xf, const float* __restrict__ fcw,
    const float* __restrict__ fcb, float* __restrict__ out)
{
  __shared__ float r1[8];
  int b = blockIdx.x, t = threadIdx.x;
  int wv = t >> 5, lane = t & 31;
  float s = 0.f;
  for (int sp = 0; sp < S_LEN; ++sp)
    s += xf[((size_t)(b * S_LEN + sp)) * D_MODEL + t];
  float v = (s / (float)S_LEN) * fcw[t];
  v = wred32(v);
  if (lane == 0) r1[wv] = v;
  __syncthreads();
  if (t == 0) {
    float tot = 0.f;
#pragma unroll
    for (int i = 0; i < 8; ++i) tot += r1[i];
    out[b] = tot + fcb[0];
  }
}

// ---------------------------------------------------------------------------
extern "C" void kernel_launch(void* const* d_in, const int* in_sizes, int n_in,
                              void* d_out, int out_size, void* d_ws, size_t ws_size,
                              hipStream_t stream) {
  (void)in_sizes; (void)n_in; (void)out_size; (void)ws_size;
  const int*   seq     = (const int*)  d_in[0];
  const float* embed   = (const float*)d_in[1];
  const float* m_ln_w  = (const float*)d_in[2];
  const float* m_Wup   = (const float*)d_in[3];
  const float* m_bup   = (const float*)d_in[4];
  const float* m_convw = (const float*)d_in[5];
  const float* m_convb = (const float*)d_in[6];
  const float* m_Wq    = (const float*)d_in[7];
  const float* m_Wk    = (const float*)d_in[8];
  const float* m_Wv    = (const float*)d_in[9];
  const float* m_Wig   = (const float*)d_in[10];
  const float* m_big   = (const float*)d_in[11];
  const float* m_Wfg   = (const float*)d_in[12];
  const float* m_bfg   = (const float*)d_in[13];
  const float* m_skip  = (const float*)d_in[14];
  const float* m_mhn_w = (const float*)d_in[15];
  const float* m_Wdown = (const float*)d_in[16];
  const float* m_bdown = (const float*)d_in[17];
  const float* s_ln_w  = (const float*)d_in[18];
  const float* s_convw = (const float*)d_in[19];
  const float* s_convb = (const float*)d_in[20];
  const float* s_Wi    = (const float*)d_in[21];
  const float* s_Wf    = (const float*)d_in[22];
  const float* s_Wz    = (const float*)d_in[23];
  const float* s_Wo    = (const float*)d_in[24];
  const float* s_Ri    = (const float*)d_in[25];
  const float* s_Rf    = (const float*)d_in[26];
  const float* s_Rz    = (const float*)d_in[27];
  const float* s_Ro    = (const float*)d_in[28];
  const float* s_bi    = (const float*)d_in[29];
  const float* s_bf    = (const float*)d_in[30];
  const float* s_bz    = (const float*)d_in[31];
  const float* s_bo    = (const float*)d_in[32];
  const float* s_gn_w  = (const float*)d_in[33];
  const float* s_ln2_w = (const float*)d_in[34];
  const float* f_Wg    = (const float*)d_in[35];
  const float* f_bg    = (const float*)d_in[36];
  const float* f_Wu    = (const float*)d_in[37];
  const float* f_bu    = (const float*)d_in[38];
  const float* f_Wd    = (const float*)d_in[39];
  const float* f_bd    = (const float*)d_in[40];
  const float* post_ln = (const float*)d_in[41];
  const float* fc_w    = (const float*)d_in[42];
  const float* fc_b    = (const float*)d_in[43];
  float* outp = (float*)d_out;

  // --- workspace bump allocator ---
  size_t off = 0;
  auto alloc = [&](size_t bytes) -> char* {
    off = (off + 255) & ~(size_t)255;
    char* p = (char*)d_ws + off;
    off += bytes;
    return p;
  };
  const size_t R256 = (size_t)NROWS * D_MODEL * 4;
  const size_t R512 = (size_t)NROWS * DI * 4;
  float* x     = (float*)alloc(R256);
  float* xln   = (float*)alloc(R256);
  float* up    = (float*)alloc((size_t)NROWS * 2 * DI * 4);
  float* xc    = (float*)alloc(R512);
  float* q     = (float*)alloc(R512);
  float* k     = (float*)alloc(R512);
  float* v     = (float*)alloc(R512);
  unsigned short* qb  = (unsigned short*)alloc((size_t)NROWS * DI * 2);
  unsigned short* kb  = (unsigned short*)alloc((size_t)NROWS * DI * 2);
  unsigned short* vT  = (unsigned short*)alloc((size_t)BATCH * NHM * DHM * SPAD * 2);
  float* ipre  = (float*)alloc((size_t)BATCH * NHM * S_LEN * 4);
  float* fpre  = (float*)alloc((size_t)BATCH * NHM * S_LEN * 4);
  float* cum   = (float*)alloc((size_t)BATCH * NHM * S_LEN * 4);
  float* gk    = (float*)alloc((size_t)BATCH * NHM * S_LEN * 4);
  float* pm    = (float*)alloc((size_t)BATCH * NHM * S_LEN * 4);
  float* hm    = (float*)alloc(R512);
  float* hgate = (float*)alloc(R512);
  float* x2    = (float*)alloc(R256);
  float* xln2  = (float*)alloc(R256);
  float* xcs   = (float*)alloc(R256);
  float* gib   = (float*)alloc(R256);
  float* gfb   = (float*)alloc(R256);
  float* gzb   = (float*)alloc(R256);
  float* gob   = (float*)alloc(R256);
  float* hss   = (float*)alloc(R256);
  float* x3    = (float*)alloc(R256);
  float* y     = (float*)alloc(R256);
  float* gbuf  = (float*)alloc((size_t)NROWS * FF_DIM * 4);
  float* ubuf  = (float*)alloc((size_t)NROWS * FF_DIM * 4);
  float* hx    = (float*)alloc((size_t)NROWS * FF_DIM * 4);
  float* x4    = (float*)alloc(R256);
  float* xf    = (float*)alloc(R256);

  auto gemm = [&](const float* A, int lda, const float* Bp, int ldb,
                  const float* bias, const float* res, float* C, int ldc,
                  int M, int N, int K) {
    dim3 grid((N + 63) / 64, (M + 63) / 64);
    gemm_bf16_kernel<<<grid, dim3(128), 0, stream>>>(A, lda, Bp, ldb, bias, res,
                                                     C, ldc, M, N, K);
  };

  // ---- block 0: mLSTM ----
  embed_ln_kernel<<<NROWS, 256, 0, stream>>>(seq, embed, m_ln_w, x, xln);
  gemm(xln, D_MODEL, m_Wup, 2 * DI, m_bup, nullptr, up, 2 * DI, NROWS, 2 * DI, D_MODEL);
  mconv_kernel<<<(NROWS * DI + 255) / 256, 256, 0, stream>>>(up, m_convw, m_convb, xc);
  qkv_kernel<<<(NROWS * 128 + 255) / 256, 256, 0, stream>>>(xc, up, m_Wq, m_Wk, m_Wv,
                                                            q, k, v, qb, kb);
  vtrans_kernel<<<(BATCH * NHM * DHM * SPAD + 255) / 256, 256, 0, stream>>>(v, vT);
  gatepre_kernel<<<NROWS, 256, 0, stream>>>(q, k, v, m_Wig, m_big, m_Wfg, m_bfg, ipre, fpre);
  decay_scan_kernel<<<1, 32, 0, stream>>>(ipre, fpre, cum, gk, pm);
  mlstm_attn_kernel<<<BATCH * NHM * NTQ, 32, 0, stream>>>(qb, kb, vT, gk, pm, cum, hm);
  mlstm_out_kernel<<<NROWS, 128, 0, stream>>>(hm, m_mhn_w, m_skip, xc, up, hgate);
  gemm(hgate, DI, m_Wdown, D_MODEL, m_bdown, x, x2, D_MODEL, NROWS, D_MODEL, DI);

  // ---- block 1: sLSTM ----
  ln256_kernel<<<NROWS, 256, 0, stream>>>(x2, s_ln_w, xln2);
  sconv_kernel<<<(NROWS * D_MODEL + 255) / 256, 256, 0, stream>>>(xln2, s_convw, s_convb, xcs);
  for (int h = 0; h < NHM; ++h) {
    gemm(xcs  + h * 64, D_MODEL, s_Wi + (size_t)h * 4096, 64, s_bi + h * 64, nullptr,
         gib + h * 64, D_MODEL, NROWS, 64, 64);
    gemm(xcs  + h * 64, D_MODEL, s_Wf + (size_t)h * 4096, 64, s_bf + h * 64, nullptr,
         gfb + h * 64, D_MODEL, NROWS, 64, 64);
    gemm(xln2 + h * 64, D_MODEL, s_Wz + (size_t)h * 4096, 64, s_bz + h * 64, nullptr,
         gzb + h * 64, D_MODEL, NROWS, 64, 64);
    gemm(xln2 + h * 64, D_MODEL, s_Wo + (size_t)h * 4096, 64, s_bo + h * 64, nullptr,
         gob + h * 64, D_MODEL, NROWS, 64, 64);
  }
  slstm_scan_kernel<<<BATCH * NHM, 64, (4 * 4096 + 64) * sizeof(float), stream>>>(
      gib, gfb, gzb, gob, s_Ri, s_Rf, s_Rz, s_Ro, hss);
  slstm_out_kernel<<<NROWS, 128, 0, stream>>>(hss, s_gn_w, x2, x3);

  // ---- FFN ----
  ln256_kernel<<<NROWS, 256, 0, stream>>>(x3, s_ln2_w, y);
  gemm(y, D_MODEL, f_Wg, FF_DIM, f_bg, nullptr, gbuf, FF_DIM, NROWS, FF_DIM, D_MODEL);
  gemm(y, D_MODEL, f_Wu, FF_DIM, f_bu, nullptr, ubuf, FF_DIM, NROWS, FF_DIM, D_MODEL);
  ffn_act_kernel<<<(NROWS * FF_DIM + 255) / 256, 256, 0, stream>>>(gbuf, ubuf, hx,
                                                                   NROWS * FF_DIM);
  gemm(hx, FF_DIM, f_Wd, D_MODEL, f_bd, x3, x4, D_MODEL, NROWS, D_MODEL, FF_DIM);

  // ---- head ----
  ln256_kernel<<<NROWS, 256, 0, stream>>>(x4, post_ln, xf);
  pool_fc_kernel<<<BATCH, 256, 0, stream>>>(xf, fc_w, fc_b, outp);
}
